// RouteNetOneToOneOutput_48902497632408
// MI455X (gfx1250) — matrix operand order repaired and checked by reading, not compiled, not run on hardware
//
#include <hip/hip_runtime.h>

typedef __attribute__((ext_vector_type(16))) _Float16 v16h;
typedef __attribute__((ext_vector_type(8)))  float    v8f;
typedef unsigned int v4u __attribute__((ext_vector_type(4)));
typedef int          v4i __attribute__((ext_vector_type(4)));
typedef int          v8i __attribute__((ext_vector_type(8)));

#define BANKS   64
#define FAN     8
#define DD      10
#define BATCH   4096
#define N_IN    784
#define BTILE   16
#define NLAYERS 15
#define N_CONN  7680

#define KT_IN 25          // ceil(784/32) K-tiles for input GEMM
#define NT_IN 40          // 640/16 N-tiles for input GEMM
#define NBT   (BATCH / BTILE)          // 256 batch tiles
#define XTILE_HALVES (KT_IN * 512)     // 12,800 halves = 25,600 B per batch tile

// d_ws layout (bytes):
//   [0]        : u32 open-gate counter
//   [128]      : Wi fragments   : NT_IN*KT_IN tiles * 512 halves   (1,024,000 B)
//   [WS_W2_OFF]: Wg/Wd fragments: NLAYERS*BANKS*6 tiles * 512 halves (5,898,240 B)
//   [WS_X_OFF] : x fragments    : NBT * XTILE_HALVES halves        (6,553,600 B)
#define WS_WI_OFF 128
#define WI_HALVES (NT_IN * KT_IN * 512)                 // 512,000
#define WS_W2_OFF (WS_WI_OFF + WI_HALVES * 2)           // 1,024,128
#define W2_HALVES (NLAYERS * BANKS * 6 * 512)           // 2,949,120
#define WS_X_OFF  (WS_W2_OFF + W2_HALVES * 2)           // 6,922,368
#define X_HALVES  (NBT * XTILE_HALVES)                  // 3,276,800
// total ws needed: ~12.9 MB

static __device__ inline v8f v8f_zero() {
  v8f z = {0.f,0.f,0.f,0.f,0.f,0.f,0.f,0.f};
  return z;
}

// ---------------------------------------------------------------------------
// Prep: swizzle Wi (fp32 [640][784]) into WMMA B-fragment layout (f16).
// B-frag layout per 32x16 tile: lane%16 = N-col, element e = K (A table order).
// Also zeroes the global open-gate counter (stream-ordered before main).
// ---------------------------------------------------------------------------
__global__ void prep_wi(const float* __restrict__ Wi, _Float16* __restrict__ wsWi,
                        unsigned int* __restrict__ cnt) {
  if (blockIdx.x == 0 && threadIdx.x == 0) *cnt = 0u;
  int idx = blockIdx.x * blockDim.x + threadIdx.x;
  if (idx >= WI_HALVES) return;
  int e    = idx & 15;
  int lane = (idx >> 4) & 31;
  int tile = idx >> 9;
  int kt   = tile % KT_IN;
  int nt   = tile / KT_IN;
  int n = nt * 16 + (lane & 15);        // output column = bank*10+d  (always < 640)
  int i = kt * 32 + ((lane >= 16) ? ((e < 8) ? e + 8 : e + 16) : ((e < 8) ? e : e + 8));     // input feature (K)
  float v = (i < N_IN) ? Wi[n * N_IN + i] : 0.0f;
  wsWi[idx] = (_Float16)v;
}

// ---------------------------------------------------------------------------
// Prep: swizzle per-(layer,bank) gate+data weights into B fragments.
// 6 N-tiles per (l,s): tile0 = gate (cols 0..7 = fan k), tiles1..5 = data
// (col c = k*10+e, 80 cols). K dim = d (0..9), zero-padded to 32.
// ---------------------------------------------------------------------------
__global__ void prep_w2(const float* __restrict__ Wg, const float* __restrict__ Wd,
                        _Float16* __restrict__ wsW2) {
  int idx = blockIdx.x * blockDim.x + threadIdx.x;
  if (idx >= W2_HALVES) return;
  int s16  = idx & 15;
  int lane = (idx >> 4) & 31;
  int e    = lane & 15;
  int tile = idx >> 9;
  int nt = tile % 6;
  int s  = (tile / 6) % BANKS;
  int l  = tile / (6 * BANKS);
  float v = 0.0f;
  int d = (lane >= 16) ? ((s16 < 8) ? s16 + 8 : s16 + 16) : ((s16 < 8) ? s16 : s16 + 8);               // K index = hidden dim
  if (d < DD) {
    if (nt == 0) {
      if (e < FAN) v = Wg[((l * BANKS + s) * FAN + e) * DD + d];
    } else {
      int c = (nt - 1) * 16 + e;
      if (c < FAN * DD) {
        int f = c / DD, eo = c % DD;
        v = Wd[(((l * BANKS + s) * FAN + f) * DD + eo) * DD + d];
      }
    }
  }
  wsW2[idx] = (_Float16)v;
}

// ---------------------------------------------------------------------------
// Prep: convert + swizzle x into per-batch-tile A-fragment blocks (f16),
// contiguous per tile so the main kernel can TDM-DMA them straight to LDS.
// A-frag (16-bit, 16x32): lane m holds K {0..7,16..23}; lane 16+m holds
// K {8..15,24..31}.
// ---------------------------------------------------------------------------
__global__ void prep_x(const float* __restrict__ x, _Float16* __restrict__ wsX) {
  int idx = blockIdx.x * blockDim.x + threadIdx.x;
  if (idx >= X_HALVES) return;
  int e    = idx & 15;
  int lane = (idx >> 4) & 31;
  int tile = idx >> 9;
  int kt = tile % KT_IN;
  int bt = tile / KT_IN;
  int m  = lane & 15;
  bool hi = lane >= 16;
  int k = hi ? ((e < 8) ? e + 8 : e + 16)
             : ((e < 8) ? e     : e + 8);
  int i = kt * 32 + k;
  float v = (i < N_IN) ? x[(bt * BTILE + m) * N_IN + i] : 0.0f;
  wsX[idx] = (_Float16)v;
}

// ---------------------------------------------------------------------------
// Main: one workgroup owns a 16-row batch tile for the WHOLE network.
// 8 waves; per layer 8 rounds; round r: wave w handles source bank s=8w+r,
// whose targets s..s+7 (mod 64) are disjoint across waves -> non-atomic,
// deterministic LDS accumulation.
// ---------------------------------------------------------------------------
__global__ __launch_bounds__(256) void routenet_main(
    const float* __restrict__ bi,
    const float* __restrict__ bg, const float* __restrict__ Wo,
    const _Float16* __restrict__ wsWi, const _Float16* __restrict__ wsW2,
    const _Float16* __restrict__ wsX,
    unsigned int* __restrict__ cnt, float* __restrict__ out) {

  __shared__ __attribute__((aligned(32))) _Float16 frag[BANKS * 512];    // 64 KB: acts A-fragments
  __shared__ __attribute__((aligned(32))) _Float16 xfrag[XTILE_HALVES];  // 25.6 KB: x A-fragments
  __shared__ float acc[BTILE * BANKS * DD];                              // 40 KB: scatter accumulator
  __shared__ float gred[256][8];                                        // gate-sum reduction
  __shared__ unsigned int nred[256];                                    // open-gate reduction

  const int tid  = threadIdx.x;
  const int lane = tid & 31;
  const int w    = tid >> 5;
  const int bt   = blockIdx.x;
  const int c16  = lane & 15;
  const bool hi  = lane >= 16;

  // ---- Phase A: zero frag/acc; TDM-DMA this tile's x fragments into LDS ----
  if (tid < 32) {
    // Wave 0 issues one Tensor Data Mover load: 3200 x 8B contiguous
    // (25,600 B) from wsX[bt] into LDS xfrag. TDM ignores EXEC and issues
    // per executing wave, so only wave 0 runs this (uniform branch).
    unsigned lds_addr = (unsigned)(uintptr_t)&xfrag[0];   // flat LDS addr low 32 = LDS offset
    unsigned long long ga =
        (unsigned long long)(uintptr_t)(wsX + (size_t)bt * XTILE_HALVES);
    const unsigned n64 = XTILE_HALVES / 4;                // 3200 8-byte elements
    v4u g0;
    g0.x = 1u;                                            // count=1 valid descriptor
    g0.y = lds_addr;                                      // lds_addr[31:0]
    g0.z = (unsigned)(ga & 0xFFFFFFFFu);                  // global_addr[31:0]
    g0.w = (unsigned)((ga >> 32) & 0x01FFFFFFu) | (2u << 30); // addr[56:32] | type=2
    v8i g1;
    g1[0] = (int)(3u << 16);                              // data_size=8B; no mask/flags
    g1[1] = (int)((n64 & 0xFFFFu) << 16);                 // tensor_dim0[15:0]
    g1[2] = (int)((1u << 16) | (n64 >> 16));              // tensor_dim1=1 | tensor_dim0[31:16]
    g1[3] = (int)((n64 & 0xFFFFu) << 16);                 // tile_dim0 | tensor_dim1[31:16]
    g1[4] = 1;                                            // tile_dim1=1, tile_dim2=0
    g1[5] = (int)n64;                                     // tensor_dim0_stride[31:0]
    g1[6] = (int)((n64 & 0xFFFFu) << 16);                 // dim1_stride[15:0] | dim0_stride[47:32]
    g1[7] = 0;                                            // dim1_stride[47:16]
    v4i gz4 = {0, 0, 0, 0};
    v8i gz8 = {0, 0, 0, 0, 0, 0, 0, 0};
    __builtin_amdgcn_tensor_load_to_lds(g0, g1, gz4, gz4, gz8, 0);
    __builtin_amdgcn_s_wait_tensorcnt(0);
  }
  for (int i = tid; i < BANKS * 512; i += 256) frag[i] = (_Float16)0.0f;
  for (int i = tid; i < BTILE * BANKS * DD; i += 256) acc[i] = 0.0f;
  __syncthreads();

  // ---- Phase B: input GEMM [16x784] x [784x640], bias + ReLU -> frag ----
  {
    v8f cacc[5];
    for (int p = 0; p < 5; ++p) cacc[p] = v8f_zero();
    for (int kt = 0; kt < KT_IN; ++kt) {
      v16h a = *(const v16h*)&xfrag[kt * 512 + lane * 16];
      #pragma unroll
      for (int p = 0; p < 5; ++p) {
        int nt = w * 5 + p;
        v16h b = *(const v16h*)&wsWi[(nt * KT_IN + kt) * 512 + lane * 16];
        cacc[p] = __builtin_amdgcn_wmma_f32_16x16x32_f16(
            false, a, false, b, (short)0, cacc[p], false, false);
      }
    }
    #pragma unroll
    for (int p = 0; p < 5; ++p) {
      int nt = w * 5 + p;
      int n = nt * 16 + c16;               // global output column, < 640
      float bias = bi[n];
      int bank = n / DD, dd = n % DD;
      int lbase = (dd < 8) ? 0 : 16;
      int e2    = (dd < 8) ? dd : dd - 8;
      #pragma unroll
      for (int rr = 0; rr < 8; ++rr) {
        int m = rr + (hi ? 8 : 0);
        float v = fmaxf(cacc[p][rr] + bias, 0.0f);
        frag[bank * 512 + (lbase + m) * 16 + e2] = (_Float16)v;
      }
    }
  }
  __syncthreads();

  // ---- Phase C: 15 routing layers ----
  float gs[8] = {0.f,0.f,0.f,0.f,0.f,0.f,0.f,0.f};
  unsigned int nopen = 0;

  for (int l = 0; l < NLAYERS; ++l) {
    for (int r = 0; r < 8; ++r) {
      int s = w * 8 + r;                                   // this wave's source bank
      v16h a = *(const v16h*)&frag[s * 512 + lane * 16];
      const _Float16* wb = &wsW2[(unsigned)((l * BANKS + s) * 6) * 512u];

      // gate GEMM (tile 0): cols 0..7 = fan index
      v16h bg0 = *(const v16h*)&wb[lane * 16];
      v8f glog = __builtin_amdgcn_wmma_f32_16x16x32_f16(
          false, a, false, bg0, (short)0, v8f_zero(), false, false);
      v8f gv = v8f_zero();
      if (c16 < FAN) {
        float bgv = bg[(l * BANKS + s) * FAN + c16];
        #pragma unroll
        for (int rr = 0; rr < 8; ++rr) {
          float lg = glog[rr] + bgv;
          float g  = fminf(fmaxf(lg, 0.0f), 1.0f);
          gv[rr] = g;
          gs[rr] += g;
          nopen  += (lg > 0.0f) ? 1u : 0u;
        }
      }

      // data GEMMs (tiles 1..5): col c = k*10+e; gate via cross-lane shuffle;
      // accumulate into exclusive target range [s..s+7] (mod 64).
      #pragma unroll
      for (int nt = 1; nt < 6; ++nt) {
        v16h bd = *(const v16h*)&wb[nt * 512 + lane * 16];
        v8f dt = __builtin_amdgcn_wmma_f32_16x16x32_f16(
            false, a, false, bd, (short)0, v8f_zero(), false, false);
        int col = (nt - 1) * 16 + c16;        // 0..79
        int k = col / DD, e = col % DD;
        int t = (s + k) & (BANKS - 1);
        int srcl = hi ? (16 + k) : k;
        #pragma unroll
        for (int rr = 0; rr < 8; ++rr) {
          float g = __shfl(gv[rr], srcl, 32);
          int m = rr + (hi ? 8 : 0);
          int ai = (m * BANKS + t) * DD + e;
          acc[ai] += g * dt[rr];              // exclusive per round -> no atomics
        }
      }
      __syncthreads();                        // hand target ownership to next round
    }

    // ReLU + rebuild acts fragments for the next layer (or finalize acts)
    if (l < NLAYERS - 1) {
      for (int idx = tid; idx < BTILE * BANKS * DD; idx += 256) {
        int m   = idx / (BANKS * DD);
        int rem = idx % (BANKS * DD);
        int t = rem / DD, d = rem % DD;
        float a2 = fmaxf(acc[idx], 0.0f);
        int ln2 = (d < 8) ? m : (16 + m);
        int e2  = (d < 8) ? d : d - 8;
        frag[t * 512 + ln2 * 16 + e2] = (_Float16)a2;
        acc[idx] = 0.0f;
      }
    } else {
      for (int idx = tid; idx < BTILE * BANKS * DD; idx += 256)
        acc[idx] = fmaxf(acc[idx], 0.0f);
    }
    __syncthreads();
  }

  // ---- Phase D: output projection per bank, ReLU ----
  for (int idx = tid; idx < BTILE * BANKS; idx += 256) {
    int m = idx / BANKS, t = idx % BANKS;
    float sum = 0.0f;
    #pragma unroll
    for (int d = 0; d < DD; ++d)
      sum += acc[(m * BANKS + t) * DD + d] * Wo[t * DD + d];
    out[(bt * BTILE + m) * BANKS + t] = fmaxf(sum, 0.0f);
  }

  // ---- Phase E: deterministic gate-stat reductions ----
  #pragma unroll
  for (int rr = 0; rr < 8; ++rr) gred[tid][rr] = gs[rr];
  nred[tid] = nopen;
  __syncthreads();
  if (tid < BTILE) {
    int m = tid;
    float tot = 0.0f;
    for (int wv = 0; wv < 8; ++wv)
      for (int li = 0; li < 8; ++li) {
        int ln = (m < 8) ? li : (16 + li);
        tot += gred[wv * 32 + ln][m & 7];
      }
    out[BATCH * BANKS + bt * BTILE + m] = tot / (float)N_CONN;
  }
  if (tid == 0) {
    unsigned int total = 0;
    for (int i = 0; i < 256; ++i) total += nred[i];
    atomicAdd(cnt, total);   // integer atomic: order-independent, deterministic
  }
}

__global__ void finalize_kernel(const unsigned int* __restrict__ cnt,
                                float* __restrict__ out) {
  if (blockIdx.x == 0 && threadIdx.x == 0)
    out[BATCH * BANKS + BATCH] = (float)(*cnt) / ((float)N_CONN * (float)BATCH);
}

extern "C" void kernel_launch(void* const* d_in, const int* in_sizes, int n_in,
                              void* d_out, int out_size, void* d_ws, size_t ws_size,
                              hipStream_t stream) {
  const float* x  = (const float*)d_in[0];
  const float* Wi = (const float*)d_in[1];
  const float* bi = (const float*)d_in[2];
  const float* Wg = (const float*)d_in[3];
  const float* bg = (const float*)d_in[4];
  const float* Wd = (const float*)d_in[5];
  const float* Wo = (const float*)d_in[6];
  float* out = (float*)d_out;

  unsigned int* cnt = (unsigned int*)d_ws;
  _Float16* wsWi = (_Float16*)((char*)d_ws + WS_WI_OFF);
  _Float16* wsW2 = (_Float16*)((char*)d_ws + WS_W2_OFF);
  _Float16* wsX  = (_Float16*)((char*)d_ws + WS_X_OFF);

  prep_wi<<<(WI_HALVES + 255) / 256, 256, 0, stream>>>(Wi, wsWi, cnt);
  prep_w2<<<(W2_HALVES + 255) / 256, 256, 0, stream>>>(Wg, Wd, wsW2);
  prep_x <<<(X_HALVES  + 255) / 256, 256, 0, stream>>>(x, wsX);
  routenet_main<<<NBT, 256, 0, stream>>>(bi, bg, Wo, wsWi, wsW2, wsX, cnt, out);
  finalize_kernel<<<1, 1, 0, stream>>>(cnt, out);
}
